// DNC_13159779795434
// MI455X (gfx1250) — matrix operand order repaired and checked by reading, not compiled, not run on hardware
//
#include <hip/hip_runtime.h>
#include <hip/hip_bf16.h>
#include <math.h>

// ---------------- problem constants ----------------
constexpr int Bc  = 16;
constexpr int Nc  = 2048;
constexpr int Mc  = 256;
constexpr int RHc = 4;
constexpr int Hc  = 1024;
constexpr int CINc = Mc + Mc * RHc;        // 1280
constexpr int RPLc = Mc + 4;               // 260
constexpr int WPLc = 3 * Mc + RHc + 3;     // 775
constexpr int H3c  = 3 * Hc;               // 3072
constexpr float EPSc = 1e-12f;

typedef __attribute__((ext_vector_type(16))) _Float16 v16h;
typedef __attribute__((ext_vector_type(8)))  float    v8f;

__device__ __forceinline__ float sigm(float x) { return 1.0f / (1.0f + __expf(-x)); }
__device__ __forceinline__ float4 ld4(const float* p) { return *reinterpret_cast<const float4*>(p); }

__device__ __forceinline__ v16h cvt16(float4 a, float4 b, float4 c, float4 d)
{
    v16h r;
    r[0]=(_Float16)a.x;  r[1]=(_Float16)a.y;  r[2]=(_Float16)a.z;  r[3]=(_Float16)a.w;
    r[4]=(_Float16)b.x;  r[5]=(_Float16)b.y;  r[6]=(_Float16)b.z;  r[7]=(_Float16)b.w;
    r[8]=(_Float16)c.x;  r[9]=(_Float16)c.y;  r[10]=(_Float16)c.z; r[11]=(_Float16)c.w;
    r[12]=(_Float16)d.x; r[13]=(_Float16)d.y; r[14]=(_Float16)d.z; r[15]=(_Float16)d.w;
    return r;
}

// =====================================================================
// Generic GEMM via V_WMMA_F32_16X16X32_F16, templated on the B element
// stride (compile-time at all call sites).
//   Out[r,c] = sum_k A[r,k] * W[c*wRow + k*WELEM] + bias[c]
// One wave computes a 16x32 output tile (two 16x16 WMMA tiles sharing the
// A fragment). WELEM==1: float4-vectorized B loads, ragged cols handled by
// redirecting OOB lanes to a zeroed pad with pointer-step 0 (branch-free,
// EXEC stays all-ones for WMMA). WELEM>1: strided B loads become
// global_load_b32 with immediate offsets (j*WELEM*4 < 2^23); cols must be
// a multiple of 32 at those call sites. blockIdx.y = batch. K % 32 == 0.
// =====================================================================
template<int WELEM>
__global__ void wmma_gemm16(const float* __restrict__ A, long long aBatch, int lda, int rowsA,
                            const float* __restrict__ W, long long wBatch, long long wRow,
                            const float* __restrict__ bias,
                            float* __restrict__ Out, long long oBatch, int ldo,
                            int cols, int K, const float* __restrict__ zeropad)
{
    const int lane = threadIdx.x;                  // wave32, full EXEC for WMMA
    const int bb   = blockIdx.y;
    A   += (long long)bb * aBatch;
    W   += (long long)bb * wBatch;
    Out += (long long)bb * oBatch;

    // ---- A fragment addressing (contiguous k order: k = 16*half + s) ----
    const int  rA    = lane & 15;
    const int  aKsel = (lane < 16) ? 0 : 16;
    const bool rowOK = (rA < rowsA);
    const float* aptr = rowOK ? (A + (long long)rA * lda + aKsel) : zeropad;
    const long long aStep = rowOK ? 32 : 0;

    // ---- B fragment addressing (contiguous k order: k = 16*half + s), two column tiles ----
    const int  bKsel = (lane < 16) ? 0 : 16;
    const int  col0  = blockIdx.x * 32 + (lane & 15);
    const int  col1  = col0 + 16;
    const bool c0 = (col0 < cols), c1 = (col1 < cols);

    v8f acc0 = {}, acc1 = {};

    if constexpr (WELEM == 1) {
        const float* bp0 = c0 ? (W + (long long)col0 * wRow + bKsel) : zeropad;
        const float* bp1 = c1 ? (W + (long long)col1 * wRow + bKsel) : zeropad;
        const long long bStep0 = c0 ? 32 : 0;
        const long long bStep1 = c1 ? 32 : 0;
        for (int k0 = 0; k0 < K; k0 += 32) {
            v16h af = cvt16(ld4(aptr), ld4(aptr + 4), ld4(aptr + 8), ld4(aptr + 12));
            aptr += aStep;
            __builtin_prefetch(bp0 + bStep0, 0, 0);        // global_prefetch_b8
            v16h bf0 = cvt16(ld4(bp0), ld4(bp0 + 4), ld4(bp0 + 8), ld4(bp0 + 12));
            bp0 += bStep0;
            v16h bf1 = cvt16(ld4(bp1), ld4(bp1 + 4), ld4(bp1 + 8), ld4(bp1 + 12));
            bp1 += bStep1;
            acc0 = __builtin_amdgcn_wmma_f32_16x16x32_f16(false, af, false, bf0,
                                                          (short)0, acc0, false, false);
            acc1 = __builtin_amdgcn_wmma_f32_16x16x32_f16(false, af, false, bf1,
                                                          (short)0, acc1, false, false);
        }
    } else {
        // cols is a multiple of 32 at all strided call sites -> no col guard
        const float* bp0 = W + (long long)col0 * wRow + (long long)bKsel * WELEM;
        const float* bp1 = W + (long long)col1 * wRow + (long long)bKsel * WELEM;
        for (int k0 = 0; k0 < K; k0 += 32) {
            v16h af = cvt16(ld4(aptr), ld4(aptr + 4), ld4(aptr + 8), ld4(aptr + 12));
            aptr += aStep;
            v16h bf0, bf1;
            #pragma unroll
            for (int j = 0; j < 16; ++j) bf0[j] = (_Float16)bp0[j * WELEM];   // imm offsets
            bp0 += 32 * WELEM;
            #pragma unroll
            for (int j = 0; j < 16; ++j) bf1[j] = (_Float16)bp1[j * WELEM];
            bp1 += 32 * WELEM;
            acc0 = __builtin_amdgcn_wmma_f32_16x16x32_f16(false, af, false, bf0,
                                                          (short)0, acc0, false, false);
            acc1 = __builtin_amdgcn_wmma_f32_16x16x32_f16(false, af, false, bf1,
                                                          (short)0, acc1, false, false);
        }
    }

    const float bv0 = (bias != nullptr && c0) ? bias[col0] : 0.f;
    const float bv1 = (bias != nullptr && c1) ? bias[col1] : 0.f;
    #pragma unroll
    for (int v = 0; v < 8; ++v) {
        const int ro = v + ((lane < 16) ? 0 : 8);   // C/D layout: VGPR v -> row v / v+8
        if (ro < rowsA) {
            if (c0) Out[(long long)ro * ldo + col0] = acc0[v] + bv0;
            if (c1) Out[(long long)ro * ldo + col1] = acc1[v] + bv1;
        }
    }
}

// =====================================================================
// Elementwise / reduction kernels
// =====================================================================
__global__ void zeropad_kernel(float* __restrict__ z)
{
    z[threadIdx.x] = 0.f;                           // 32 floats
}

__global__ void concat_kernel(const float* __restrict__ x, const float* __restrict__ rv,
                              float* __restrict__ ci)
{
    int i = blockIdx.x * 256 + threadIdx.x;         // B*CIN
    int b = i / CINc, c = i - b * CINc;
    ci[i] = (c < Mc) ? x[b * Mc + c] : rv[b * (RHc * Mc) + (c - Mc)];
}

__global__ void gru_gate_kernel(const float* __restrict__ gi, const float* __restrict__ gh,
                                const float* __restrict__ hprev, float* __restrict__ hout)
{
    int i = blockIdx.x * 256 + threadIdx.x;         // B*H
    int b = i >> 10, h = i & (Hc - 1);
    const float* gib = gi + (long long)b * H3c;
    const float* ghb = gh + (long long)b * H3c;
    float r = sigm(gib[h] + ghb[h]);
    float z = sigm(gib[Hc + h] + ghb[Hc + h]);
    float n = tanhf(gib[2 * Hc + h] + r * ghb[2 * Hc + h]);
    hout[i] = (1.f - z) * n + z * hprev[i];
}

// build normalized 5-row key set (4 read + 1 write), beta_pos, read-mode softmax
__global__ void keyprep_kernel(const float* __restrict__ rh, const float* __restrict__ wh,
                               float* __restrict__ keys5, float* __restrict__ beta5,
                               float* __restrict__ rmodes)
{
    int b5 = blockIdx.x, b = b5 / 5, r = b5 % 5;
    const float* src = (r < RHc) ? (rh + ((long long)b * RHc + r) * RPLc)
                                 : (wh + (long long)b * WPLc);
    int t = threadIdx.x;                 // 256 == Mc
    float kv = src[t];
    __shared__ float red[256];
    red[t] = kv * kv; __syncthreads();
    for (int s = 128; s > 0; s >>= 1) { if (t < s) red[t] += red[t + s]; __syncthreads(); }
    float inv = 1.0f / (sqrtf(red[0]) + EPSc);
    keys5[(long long)b5 * Mc + t] = kv * inv;
    if (t == 0) {
        float beta = src[Mc];
        float sp = (beta > 20.f) ? beta : log1pf(__expf(beta));
        beta5[b5] = sp + 1.0f;
        if (r < RHc) {
            float a = src[Mc + 1], c = src[Mc + 2], d = src[Mc + 3];
            float mx = fmaxf(a, fmaxf(c, d));
            float ea = __expf(a - mx), eb = __expf(c - mx), ec = __expf(d - mx);
            float ss = ea + eb + ec;
            float* rm = rmodes + (long long)(b * RHc + r) * 3;
            rm[0] = ea / ss; rm[1] = eb / ss; rm[2] = ec / ss;
        }
    }
}

__global__ void memnorm_kernel(const float* __restrict__ mem, float* __restrict__ invn)
{
    int row = blockIdx.x;                           // B*N rows
    const float* mr = mem + (long long)row * Mc;
    int t = threadIdx.x;                            // 64
    float s = 0.f;
    for (int i = t; i < Mc; i += 64) { float v = mr[i]; s += v * v; }
    __shared__ float red[64];
    red[t] = s; __syncthreads();
    for (int st = 32; st > 0; st >>= 1) { if (t < st) red[t] += red[t + st]; __syncthreads(); }
    if (t == 0) invn[row] = 1.0f / (sqrtf(red[0]) + EPSc);
}

// softmax over N of beta * cos * invnorm (in place on cw buffer)
__global__ void content_softmax_kernel(const float* __restrict__ beta5,
                                       const float* __restrict__ invn,
                                       float* __restrict__ cw)
{
    int b5 = blockIdx.x, b = b5 / 5;
    float bp = beta5[b5];
    float* row = cw + (long long)b5 * Nc;
    const float* inr = invn + (long long)b * Nc;
    __shared__ float red[256];
    int t = threadIdx.x;
    float lv[8];
    float mx = -INFINITY;
    #pragma unroll
    for (int i = 0; i < 8; ++i) {
        int n = t + i * 256;
        float l = bp * row[n] * inr[n];
        lv[i] = l; mx = fmaxf(mx, l);
    }
    red[t] = mx; __syncthreads();
    for (int s = 128; s > 0; s >>= 1) { if (t < s) red[t] = fmaxf(red[t], red[t + s]); __syncthreads(); }
    mx = red[0]; __syncthreads();
    float sum = 0.f;
    #pragma unroll
    for (int i = 0; i < 8; ++i) { lv[i] = __expf(lv[i] - mx); sum += lv[i]; }
    red[t] = sum; __syncthreads();
    for (int s = 128; s > 0; s >>= 1) { if (t < s) red[t] += red[t + s]; __syncthreads(); }
    float inv = 1.0f / red[0];
    #pragma unroll
    for (int i = 0; i < 8; ++i) { int n = t + i * 256; row[n] = lv[i] * inv; }
}

// final = m0*bw + m1*cw + m2*fw, then row-normalize -> new_read_w
__global__ void combine_read_w_kernel(const float* __restrict__ rmodes,
                                      const float* __restrict__ bw,
                                      const float* __restrict__ cw5,
                                      const float* __restrict__ fw,
                                      float* __restrict__ nrw)
{
    int br = blockIdx.x, b = br / RHc, r = br % RHc;
    float m0 = rmodes[br * 3 + 0], m1 = rmodes[br * 3 + 1], m2 = rmodes[br * 3 + 2];
    const float* bwr = bw + (long long)br * Nc;
    const float* fwr = fw + (long long)br * Nc;
    const float* cwr = cw5 + ((long long)b * 5 + r) * Nc;
    float* outr = nrw + (long long)br * Nc;
    int t = threadIdx.x;
    float fv[8], sum = 0.f;
    #pragma unroll
    for (int i = 0; i < 8; ++i) {
        int n = t + i * 256;
        float f = m0 * bwr[n] + m1 * cwr[n] + m2 * fwr[n];
        fv[i] = f; sum += f;
    }
    __shared__ float red[256];
    red[t] = sum; __syncthreads();
    for (int s = 128; s > 0; s >>= 1) { if (t < s) red[t] += red[t + s]; __syncthreads(); }
    float inv = 1.0f / (red[0] + EPSc);
    #pragma unroll
    for (int i = 0; i < 8; ++i) { int n = t + i * 256; outr[n] = fv[i] * inv; }
}

__global__ void usage_kernel(const float* __restrict__ wh, const float* __restrict__ nrw,
                             const float* __restrict__ usage, const float* __restrict__ pw,
                             float* __restrict__ uo)
{
    int i = blockIdx.x * 256 + threadIdx.x;         // B*N
    int b = i >> 11, n = i & (Nc - 1);
    float ret = 1.f;
    #pragma unroll
    for (int r = 0; r < RHc; ++r) {
        float fg = sigm(wh[(long long)b * WPLc + 3 * Mc + 1 + r]);
        ret *= (1.f - fg * nrw[((long long)b * RHc + r) * Nc + n]);
    }
    float uu = usage[i] + pw[i] - usage[i] * pw[i];
    uu *= ret;
    uo[i] = fminf(fmaxf(uu, 0.f), 1.f);
}

// per-batch: bitonic argsort ascending, exclusive cumprod, scatter allocation
__global__ void sort_alloc_kernel(const float* __restrict__ u, float* __restrict__ alloc)
{
    __shared__ float sv[Nc];
    __shared__ int   si[Nc];
    __shared__ float sp[Nc];
    int b = blockIdx.x, t = threadIdx.x;            // 1024 threads
    const float* ub = u + (long long)b * Nc;
    for (int i = t; i < Nc; i += 1024) { sv[i] = ub[i]; si[i] = i; }
    __syncthreads();
    for (int k = 2; k <= Nc; k <<= 1) {
        for (int j = k >> 1; j > 0; j >>= 1) {
            #pragma unroll
            for (int base = 0; base < Nc; base += 1024) {
                int i = base + t;
                int ixj = i ^ j;
                if (ixj > i) {
                    bool up = ((i & k) == 0);
                    float a = sv[i], c = sv[ixj];
                    int ia = si[i], ic = si[ixj];
                    bool sw = up ? (a > c || (a == c && ia > ic))
                                 : (a < c || (a == c && ia < ic));
                    if (sw) { sv[i] = c; sv[ixj] = a; si[i] = ic; si[ixj] = ia; }
                }
            }
            __syncthreads();
        }
    }
    for (int i = t; i < Nc; i += 1024) sp[i] = sv[i];
    __syncthreads();
    for (int off = 1; off < Nc; off <<= 1) {        // inclusive product scan
        int i0 = t, i1 = t + 1024;
        float n0 = (i0 >= off) ? sp[i0 - off] * sp[i0] : sp[i0];
        float n1 = (i1 >= off) ? sp[i1 - off] * sp[i1] : sp[i1];
        __syncthreads();
        sp[i0] = n0; sp[i1] = n1;
        __syncthreads();
    }
    for (int i = t; i < Nc; i += 1024) {
        float cp = (i == 0) ? 1.0f : sp[i - 1];     // exclusive
        alloc[(long long)b * Nc + si[i]] = (1.0f - sv[i]) * cp;
    }
}

__global__ void write_w_kernel(const float* __restrict__ wh, const float* __restrict__ alloc,
                               const float* __restrict__ cw5, float* __restrict__ ww,
                               float* __restrict__ wwsum)
{
    int b = blockIdx.x, t = threadIdx.x;            // 256 threads
    float ga = sigm(wh[(long long)b * WPLc + 3 * Mc + 1 + RHc]);
    float gw = sigm(wh[(long long)b * WPLc + 3 * Mc + 2 + RHc]);
    const float* cwr = cw5 + ((long long)b * 5 + 4) * Nc;
    float sum = 0.f;
    #pragma unroll
    for (int i = 0; i < 8; ++i) {
        int n = t + i * 256;
        float w = gw * (ga * alloc[(long long)b * Nc + n] + (1.f - ga) * cwr[n]);
        ww[(long long)b * Nc + n] = w; sum += w;
    }
    __shared__ float red[256];
    red[t] = sum; __syncthreads();
    for (int s = 128; s > 0; s >>= 1) { if (t < s) red[t] += red[t + s]; __syncthreads(); }
    if (t == 0) wwsum[b] = red[0];
}

__global__ void precedence_kernel(const float* __restrict__ prec, const float* __restrict__ ww,
                                  const float* __restrict__ wwsum, float* __restrict__ po)
{
    int i = blockIdx.x * 256 + threadIdx.x;         // B*N
    int b = i >> 11;
    po[i] = (1.f - wwsum[b]) * prec[i] + ww[i];
}

__global__ void link_kernel(const float* __restrict__ link, const float* __restrict__ ww,
                            const float* __restrict__ prec, float* __restrict__ Lout)
{
    int bi = blockIdx.x;                            // b*N + i
    int b = bi >> 11, i = bi & (Nc - 1);
    float wwi = ww[bi];
    const float4* lr = (const float4*)(link + (long long)bi * Nc);
    const float4* wj = (const float4*)(ww + (long long)b * Nc);
    const float4* pj = (const float4*)(prec + (long long)b * Nc);
    float4* od = (float4*)(Lout + (long long)bi * Nc);
    for (int q = threadIdx.x; q < Nc / 4; q += 256) {
        float4 l = lr[q], w = wj[q], p = pj[q], o;
        int j0 = q * 4;
        o.x = (i == j0 + 0) ? 0.f : ((1.f - wwi - w.x) * l.x + wwi * p.x);
        o.y = (i == j0 + 1) ? 0.f : ((1.f - wwi - w.y) * l.y + wwi * p.y);
        o.z = (i == j0 + 2) ? 0.f : ((1.f - wwi - w.z) * l.z + wwi * p.z);
        o.w = (i == j0 + 3) ? 0.f : ((1.f - wwi - w.w) * l.w + wwi * p.w);
        od[q] = o;
    }
}

__global__ void mem_update_kernel(const float* __restrict__ mem, const float* __restrict__ ww,
                                  const float* __restrict__ wh, float* __restrict__ out)
{
    long long i4 = (long long)blockIdx.x * 256 + threadIdx.x;   // B*N*M/4 float4s
    long long flat = i4 * 4;
    int b = (int)(flat / ((long long)Nc * Mc));
    long long rem = flat - (long long)b * Nc * Mc;
    int n = (int)(rem >> 8);
    int m = (int)(rem & (Mc - 1));
    float w = ww[(long long)b * Nc + n];
    const float* whb = wh + (long long)b * WPLc;
    float4 mv = ld4(mem + flat), o;
    o.x = mv.x * (1.f - w * sigm(whb[Mc + 1 + m + 0])) + w * whb[2 * Mc + 1 + m + 0];
    o.y = mv.y * (1.f - w * sigm(whb[Mc + 1 + m + 1])) + w * whb[2 * Mc + 1 + m + 1];
    o.z = mv.z * (1.f - w * sigm(whb[Mc + 1 + m + 2])) + w * whb[2 * Mc + 1 + m + 2];
    o.w = mv.w * (1.f - w * sigm(whb[Mc + 1 + m + 3])) + w * whb[2 * Mc + 1 + m + 3];
    *reinterpret_cast<float4*>(out + flat) = o;
}

// =====================================================================
// Host orchestration
// =====================================================================
extern "C" void kernel_launch(void* const* d_in, const int* in_sizes, int n_in,
                              void* d_out, int out_size, void* d_ws, size_t ws_size,
                              hipStream_t stream)
{
    const float* x       = (const float*)d_in[0];
    const float* rvecs   = (const float*)d_in[1];
    const float* memory  = (const float*)d_in[2];
    const float* link    = (const float*)d_in[3];
    const float* prec    = (const float*)d_in[4];
    const float* usage   = (const float*)d_in[5];
    const float* read_w  = (const float*)d_in[6];
    const float* pww     = (const float*)d_in[7];
    const float* hidden  = (const float*)d_in[8];
    const float* w_ih0   = (const float*)d_in[9];
    const float* w_hh0   = (const float*)d_in[10];
    const float* b_ih0   = (const float*)d_in[11];
    const float* b_hh0   = (const float*)d_in[12];
    const float* w_ih1   = (const float*)d_in[13];
    const float* w_hh1   = (const float*)d_in[14];
    const float* b_ih1   = (const float*)d_in[15];
    const float* b_hh1   = (const float*)d_in[16];
    const float* W_read  = (const float*)d_in[17];
    const float* b_read  = (const float*)d_in[18];
    const float* W_write = (const float*)d_in[19];
    const float* b_write = (const float*)d_in[20];
    const float* W_out   = (const float*)d_in[21];
    const float* b_out   = (const float*)d_in[22];

    float* out = (float*)d_out;
    // output layout (reference tuple order, flat)
    long long o_y    = 0;
    long long o_mem  = o_y    + (long long)Bc * Mc;
    long long o_L    = o_mem  + (long long)Bc * Nc * Mc;
    long long o_u    = o_L    + (long long)Bc * Nc * Nc;
    long long o_prec = o_u    + (long long)Bc * Nc;
    long long o_nrw  = o_prec + (long long)Bc * Nc;
    long long o_rv   = o_nrw  + (long long)Bc * RHc * Nc;
    long long o_ww   = o_rv   + (long long)Bc * RHc * Mc;
    long long o_hid  = o_ww   + (long long)Bc * Nc;

    float* out_y    = out + o_y;
    float* out_mem  = out + o_mem;
    float* out_L    = out + o_L;
    float* out_u    = out + o_u;
    float* out_prec = out + o_prec;
    float* out_nrw  = out + o_nrw;
    float* out_rv   = out + o_rv;
    float* out_ww   = out + o_ww;
    float* out_h0   = out + o_hid;
    float* out_h1   = out + o_hid + (long long)Bc * Hc;

    // workspace layout
    float* ws = (float*)d_ws;
    long long p = 0;
    float* zpad   = ws + p; p += 32;
    float* ci     = ws + p; p += (long long)Bc * CINc;
    float* gi0    = ws + p; p += (long long)Bc * H3c;
    float* gh0    = ws + p; p += (long long)Bc * H3c;
    float* gi1    = ws + p; p += (long long)Bc * H3c;
    float* gh1    = ws + p; p += (long long)Bc * H3c;
    float* rh_raw = ws + p; p += (long long)Bc * RHc * RPLc;
    float* wh_raw = ws + p; p += (long long)Bc * WPLc; p = (p + 3) & ~3LL;
    float* keys5  = ws + p; p += (long long)Bc * 5 * Mc;
    float* beta5  = ws + p; p += (long long)Bc * 5;
    float* rmodes = ws + p; p += (long long)Bc * RHc * 3;
    float* invn   = ws + p; p += (long long)Bc * Nc;
    float* cw5    = ws + p; p += (long long)Bc * 5 * Nc;
    float* fw     = ws + p; p += (long long)Bc * RHc * Nc;
    float* bw     = ws + p; p += (long long)Bc * RHc * Nc;
    float* allocb = ws + p; p += (long long)Bc * Nc;
    float* wwsum  = ws + p; p += Bc;
    (void)ws_size; (void)in_sizes; (void)n_in; (void)out_size;

    const float* hid0 = hidden;
    const float* hid1 = hidden + (long long)Bc * Hc;

    // ---- zero pad buffer for branch-free OOB fragment loads
    zeropad_kernel<<<1, 32, 0, stream>>>(zpad);

    // ---- input concat: ci = [x, read_vecs]
    concat_kernel<<<(Bc * CINc) / 256, 256, 0, stream>>>(x, rvecs, ci);

    // ---- GRU layer 0
    wmma_gemm16<1><<<dim3(H3c / 32, 1), 32, 0, stream>>>(ci, 0, CINc, Bc, w_ih0, 0, CINc,
                                                         b_ih0, gi0, 0, H3c, H3c, CINc, zpad);
    wmma_gemm16<1><<<dim3(H3c / 32, 1), 32, 0, stream>>>(hid0, 0, Hc, Bc, w_hh0, 0, Hc,
                                                         b_hh0, gh0, 0, H3c, H3c, Hc, zpad);
    gru_gate_kernel<<<(Bc * Hc) / 256, 256, 0, stream>>>(gi0, gh0, hid0, out_h0);

    // ---- GRU layer 1
    wmma_gemm16<1><<<dim3(H3c / 32, 1), 32, 0, stream>>>(out_h0, 0, Hc, Bc, w_ih1, 0, Hc,
                                                         b_ih1, gi1, 0, H3c, H3c, Hc, zpad);
    wmma_gemm16<1><<<dim3(H3c / 32, 1), 32, 0, stream>>>(hid1, 0, Hc, Bc, w_hh1, 0, Hc,
                                                         b_hh1, gh1, 0, H3c, H3c, Hc, zpad);
    gru_gate_kernel<<<(Bc * Hc) / 256, 256, 0, stream>>>(gi1, gh1, hid1, out_h1);

    // ---- heads: read / write / output projections
    const int RCOLS = RHc * RPLc;   // 1040
    wmma_gemm16<1><<<dim3((RCOLS + 31) / 32, 1), 32, 0, stream>>>(out_h1, 0, Hc, Bc, W_read, 0, Hc,
                                                                  b_read, rh_raw, 0, RCOLS, RCOLS, Hc, zpad);
    wmma_gemm16<1><<<dim3((WPLc + 31) / 32, 1), 32, 0, stream>>>(out_h1, 0, Hc, Bc, W_write, 0, Hc,
                                                                 b_write, wh_raw, 0, WPLc, WPLc, Hc, zpad);
    wmma_gemm16<1><<<dim3(Mc / 32, 1), 32, 0, stream>>>(out_h1, 0, Hc, Bc, W_out, 0, Hc,
                                                        b_out, out_y, 0, Mc, Mc, Hc, zpad);

    // ---- content addressing (4 read keys + write key as 5-row batch)
    keyprep_kernel<<<Bc * 5, 256, 0, stream>>>(rh_raw, wh_raw, keys5, beta5, rmodes);
    memnorm_kernel<<<Bc * Nc, 64, 0, stream>>>(memory, invn);
    // cos[b5, n] = keys5[b5,:] . memory[b, n, :]
    wmma_gemm16<1><<<dim3(Nc / 32, Bc), 32, 0, stream>>>(keys5, 5LL * Mc, Mc, 5,
                                                         memory, (long long)Nc * Mc, Mc,
                                                         nullptr, cw5, 5LL * Nc, Nc, Nc, Mc, zpad);
    content_softmax_kernel<<<Bc * 5, 256, 0, stream>>>(beta5, invn, cw5);

    // ---- temporal link reads: fw = read_w . link^T ; bw = read_w . link
    wmma_gemm16<1><<<dim3(Nc / 32, Bc), 32, 0, stream>>>(read_w, (long long)RHc * Nc, Nc, RHc,
                                                         link, (long long)Nc * Nc, Nc,
                                                         nullptr, fw, (long long)RHc * Nc, Nc, Nc, Nc, zpad);
    wmma_gemm16<Nc><<<dim3(Nc / 32, Bc), 32, 0, stream>>>(read_w, (long long)RHc * Nc, Nc, RHc,
                                                          link, (long long)Nc * Nc, 1,
                                                          nullptr, bw, (long long)RHc * Nc, Nc, Nc, Nc, zpad);

    // ---- combine modes -> new_read_w (normalized)
    combine_read_w_kernel<<<Bc * RHc, 256, 0, stream>>>(rmodes, bw, cw5, fw, out_nrw);

    // ---- read vectors: rv[b,r,m] = sum_n nrw[b,r,n] * memory[b,n,m]
    wmma_gemm16<Mc><<<dim3(Mc / 32, Bc), 32, 0, stream>>>(out_nrw, (long long)RHc * Nc, Nc, RHc,
                                                          memory, (long long)Nc * Mc, 1,
                                                          nullptr, out_rv, (long long)RHc * Mc, Mc, Mc, Nc, zpad);

    // ---- usage, allocation, write weights
    usage_kernel<<<(Bc * Nc) / 256, 256, 0, stream>>>(wh_raw, out_nrw, usage, pww, out_u);
    sort_alloc_kernel<<<Bc, 1024, 0, stream>>>(out_u, allocb);
    write_w_kernel<<<Bc, 256, 0, stream>>>(wh_raw, allocb, cw5, out_ww, wwsum);

    // ---- link matrix + precedence + memory updates
    precedence_kernel<<<(Bc * Nc) / 256, 256, 0, stream>>>(prec, out_ww, wwsum, out_prec);
    link_kernel<<<Bc * Nc, 256, 0, stream>>>(link, out_ww, prec, out_L);
    mem_update_kernel<<<(int)(((long long)Bc * Nc * Mc / 4) / 256), 256, 0, stream>>>(
        memory, out_ww, wh_raw, out_mem);
}
